// RGCNSparseTIRComposableLayer_58411555226291
// MI455X (gfx1250) — matrix operand order repaired and checked by reading, not compiled, not run on hardware
//
#include <hip/hip_runtime.h>

// ---------------------------------------------------------------------------
// RGCN hetero forward for MI455X (gfx1250, wave32, WMMA + async LDS copy).
//
//   xw[r,n,o] = sum_f feat[n,f] * W[r,o,f]        (bf16x3-split WMMA, fp32 acc)
//   y[d,o]   += xw[type(e), src(e), o]            (fp32 HW atomics)
//
// Memory-bound: ~950 MB total traffic -> ~40us at 23.3 TB/s. feat is read
// exactly once (block iterates all 8 relations with A-fragments held in
// registers); W's bf16 hi/lo split is precomputed once and async-copied
// (GLOBAL_LOAD_ASYNC_TO_LDS_B128) into LDS per relation.
// Requires ws_size >= 204.8 MB + 272 KB.
// ---------------------------------------------------------------------------

#define N_NODES_C  100000
#define N_EDGES_C  1638400
#define IN_FEAT_C  128
#define OUT_FEAT_C 64
#define NUM_RELS_C 8

typedef __attribute__((ext_vector_type(16))) __bf16 v16bf;   // 32 B
typedef __attribute__((ext_vector_type(8)))  __bf16 v8bf;    // 16 B
typedef __attribute__((ext_vector_type(8)))  float  v8f;

// Padded bf16 row stride for W planes: 136 elems = 272 B = 17*16 B
//  -> every row base is 16B-aligned (v8bf loads ok) and the dword-bank
//     stride is 4/row, so the 16 lanes of a B-fragment hit distinct banks.
#define LDW 136
#define WPLANE_ELEMS (OUT_FEAT_C * LDW)          // 8704 bf16 per relation
#define WPLANE_BYTES (WPLANE_ELEMS * 2)          // 17408 B (divisible by 16)
#define XW_BYTES ((size_t)NUM_RELS_C * N_NODES_C * OUT_FEAT_C * 4)  // 204.8 MB

// ---------------------------------------------------------------------------
// Kernel 0: one-time bf16 hi/lo split of W, stored in padded LDS-plane layout
// [r][o][f(padded to 136)] so blocks can bulk-copy it with async b128 ops.
// ---------------------------------------------------------------------------
__global__ void rgcn_wsplit(const float* __restrict__ W,
                            __bf16* __restrict__ whi,
                            __bf16* __restrict__ wlo)
{
  int i = blockIdx.x * blockDim.x + threadIdx.x;   // over 8*64*136
  if (i >= NUM_RELS_C * WPLANE_ELEMS) return;
  int f  = i % LDW;
  int ro = i / LDW;                                // r*64 + o
  float w = (f < IN_FEAT_C) ? W[(size_t)ro * IN_FEAT_C + f] : 0.f;
  __bf16 hi = (__bf16)w;                           // v_cvt_pk_bf16_f32
  whi[i] = hi;
  wlo[i] = (__bf16)(w - (float)hi);
}

// ---------------------------------------------------------------------------
// Kernel 1: xw[r] = feat @ W[r]^T via V_WMMA_F32_16X16X32_BF16, bf16x3 split.
// Block = 256 threads = 8 waves; each wave owns one 16-row M tile, all 4
// 16-wide N tiles (OUT=64), and loops over all 8 relations with the A
// fragments (bf16 hi/lo for all 4 K-steps) cached in registers. W planes are
// staged into LDS with async global->LDS b128 copies (ASYNCcnt-tracked).
// ---------------------------------------------------------------------------
__global__ __launch_bounds__(256)
void rgcn_xw_wmma(const float* __restrict__ feat,
                  const __bf16* __restrict__ whi,
                  const __bf16* __restrict__ wlo,
                  float* __restrict__ xw)
{
  // Dynamic LDS so byte offsets for the async-copy instruction start at 0.
  extern __shared__ __align__(16) __bf16 smem[];   // [2 * WPLANE_ELEMS]
  __bf16* sWhi = smem;                             // LDS bytes [0, 17408)
  __bf16* sWlo = smem + WPLANE_ELEMS;              // LDS bytes [17408, 34816)

  const int tid  = threadIdx.x;
  const int wave = tid >> 5;
  const int lane = tid & 31;
  const int kh   = lane >> 4;    // lane half: 0 or 1
  const int lm   = lane & 15;
  const int m0   = (blockIdx.x * 8 + wave) * 16;   // 100000 = 6250 * 16 exact
  const bool active = (m0 < N_NODES_C);            // whole-tile guard

  // ---- Load + split-convert the A tile ONCE (reused for all 8 relations).
  // A 16x32 bf16 layout: lane(half=kh,m=lm) holds K = 8*kh+(0..7) and
  // K = 16+8*kh+(0..7) -> two contiguous 8-float chunks of the feat row.
  v16bf a_hi[4], a_lo[4];
  if (active) {
    const float4* rowp = (const float4*)(feat + (size_t)(m0 + lm) * IN_FEAT_C);
    #pragma unroll
    for (int ks = 0; ks < 4; ++ks) {
      float af[16];
      float4 c0 = rowp[ks * 8 + 2 * kh];
      float4 c1 = rowp[ks * 8 + 2 * kh + 1];
      float4 c2 = rowp[ks * 8 + 4 + 2 * kh];
      float4 c3 = rowp[ks * 8 + 4 + 2 * kh + 1];
      af[0]=c0.x;  af[1]=c0.y;  af[2]=c0.z;  af[3]=c0.w;
      af[4]=c1.x;  af[5]=c1.y;  af[6]=c1.z;  af[7]=c1.w;
      af[8]=c2.x;  af[9]=c2.y;  af[10]=c2.z; af[11]=c2.w;
      af[12]=c3.x; af[13]=c3.y; af[14]=c3.z; af[15]=c3.w;
      #pragma unroll
      for (int e = 0; e < 16; ++e) {
        __bf16 h = (__bf16)af[e];
        a_hi[ks][e] = h;
        a_lo[ks][e] = (__bf16)(af[e] - (float)h);
      }
    }
  }

  // ---- Loop over relations; async-stage the prebuilt W[r] planes each pass.
  for (int r = 0; r < NUM_RELS_C; ++r) {
    __syncthreads();   // previous iteration's readers are done with sW*

    // Copy 2 x 17408 B (1088 x 16B chunks per plane) global -> LDS directly.
    // GVS mode: vdst = per-lane LDS byte offset, vaddr = per-lane 32-bit
    // global byte offset, saddr = 64-bit base SGPR pair.
    for (int c = tid; c < WPLANE_BYTES / 16; c += 256) {
      unsigned lds_hi = (unsigned)(c * 16);
      unsigned lds_lo = (unsigned)(WPLANE_BYTES + c * 16);
      unsigned goff   = (unsigned)(r * WPLANE_BYTES + c * 16);
      asm volatile("global_load_async_to_lds_b128 %0, %1, %2"
                   :: "v"(lds_hi), "v"(goff), "s"((const void*)whi)
                   : "memory");
      asm volatile("global_load_async_to_lds_b128 %0, %1, %2"
                   :: "v"(lds_lo), "v"(goff), "s"((const void*)wlo)
                   : "memory");
    }
    asm volatile("s_wait_asynccnt 0x0" ::: "memory");
    __syncthreads();

    if (active) {
      v8f acc[4] = {v8f{}, v8f{}, v8f{}, v8f{}};
      #pragma unroll
      for (int ks = 0; ks < 4; ++ks) {               // K = 128, 4 steps of 32
        #pragma unroll
        for (int nt = 0; nt < 4; ++nt) {             // OUT = 64, 4 N tiles
          // B 32x16 bf16 layout: lane(n=lm,half=kh) holds K = 8*kh+(0..7) and
          // K = 16+8*kh+(0..7), i.e. B[k][o] = W[o][k] -> two 8-bf16 chunks of W row (nt*16+lm).
          const int off = (nt * 16 + lm) * LDW + ks * 32 + kh * 8;
          v8bf bh0 = *(const v8bf*)(sWhi + off);
          v8bf bh1 = *(const v8bf*)(sWhi + off + 16);
          v8bf bl0 = *(const v8bf*)(sWlo + off);
          v8bf bl1 = *(const v8bf*)(sWlo + off + 16);
          v16bf b_hi = __builtin_shufflevector(bh0, bh1, 0,1,2,3,4,5,6,7,
                                                         8,9,10,11,12,13,14,15);
          v16bf b_lo = __builtin_shufflevector(bl0, bl1, 0,1,2,3,4,5,6,7,
                                                         8,9,10,11,12,13,14,15);
          // bf16x3: A*B ~= Ahi*Bhi + Alo*Bhi + Ahi*Blo (~2^-16 rel. error)
          acc[nt] = __builtin_amdgcn_wmma_f32_16x16x32_bf16(
              false, a_hi[ks], false, b_hi, (short)0, acc[nt], false, false);
          acc[nt] = __builtin_amdgcn_wmma_f32_16x16x32_bf16(
              false, a_lo[ks], false, b_hi, (short)0, acc[nt], false, false);
          acc[nt] = __builtin_amdgcn_wmma_f32_16x16x32_bf16(
              false, a_hi[ks], false, b_lo, (short)0, acc[nt], false, false);
        }
      }

      // D layout: VGPR v -> M = v + 8*kh, N = lm. Coalesced per half-wave.
      float* out = xw + (size_t)r * N_NODES_C * OUT_FEAT_C;
      #pragma unroll
      for (int v = 0; v < 8; ++v) {
        float* yrow = out + (size_t)(m0 + v + 8 * kh) * OUT_FEAT_C + lm;
        yrow[0]  = acc[0][v];
        yrow[16] = acc[1][v];
        yrow[32] = acc[2][v];
        yrow[48] = acc[3][v];
      }
    }
  }
}

// ---------------------------------------------------------------------------
// Kernel 2: zero the output accumulator (d_out is poisoned by the harness).
// ---------------------------------------------------------------------------
__global__ void rgcn_zero(float4* __restrict__ y, int n4)
{
  int i = blockIdx.x * blockDim.x + threadIdx.x;
  if (i < n4) y[i] = make_float4(0.f, 0.f, 0.f, 0.f);
}

// ---------------------------------------------------------------------------
// Kernel 3: per-edge gather + scatter-add. 64 threads per edge (one output
// feature each): coalesced 256B gather per edge, HW fp32 global atomics.
// ---------------------------------------------------------------------------
__global__ __launch_bounds__(256)
void rgcn_scatter(const float* __restrict__ xw,
                  const int* __restrict__ esrc,
                  const int* __restrict__ edst,
                  const int* __restrict__ etyp,
                  float* __restrict__ y)
{
  unsigned long long g =
      (unsigned long long)blockIdx.x * blockDim.x + threadIdx.x;
  unsigned e = (unsigned)(g >> 6);
  unsigned o = (unsigned)(g & 63u);
  if (e >= N_EDGES_C) return;
  int s = esrc[e];
  int d = edst[e];
  int t = etyp[e];
  float m = xw[((size_t)t * N_NODES_C + (size_t)s) * OUT_FEAT_C + o];
  // relaxed agent-scope fadd -> GLOBAL_ATOMIC_ADD_F32 (no CAS loop)
  __hip_atomic_fetch_add(y + (size_t)d * OUT_FEAT_C + o, m,
                         __ATOMIC_RELAXED, __HIP_MEMORY_SCOPE_AGENT);
}

// ---------------------------------------------------------------------------
extern "C" void kernel_launch(void* const* d_in, const int* in_sizes, int n_in,
                              void* d_out, int out_size, void* d_ws, size_t ws_size,
                              hipStream_t stream)
{
  const float* feat = (const float*)d_in[0];   // [100000,128] fp32
  const float* W    = (const float*)d_in[1];   // [8,64,128]   fp32
  const int*   esrc = (const int*)d_in[2];     // [E] int32
  const int*   edst = (const int*)d_in[3];     // [E] int32
  const int*   etyp = (const int*)d_in[4];     // [E] int32
  float*       y    = (float*)d_out;           // [100000,64]  fp32

  char*   wsb = (char*)d_ws;
  float*  xw  = (float*)wsb;                              // 204.8 MB
  __bf16* whi = (__bf16*)(wsb + XW_BYTES);                // 8 planes, 139 KB
  __bf16* wlo = whi + NUM_RELS_C * WPLANE_ELEMS;          // 8 planes, 139 KB

  (void)in_sizes; (void)n_in; (void)out_size; (void)ws_size;

  // One-time W split into padded bf16 planes.
  const int nsplit = NUM_RELS_C * WPLANE_ELEMS;           // 69632
  rgcn_wsplit<<<(nsplit + 255) / 256, 256, 0, stream>>>(W, whi, wlo);

  // 6250 M tiles of 16 rows; 8 tiles per block -> 782 blocks; relations
  // are looped inside the block so feat is read exactly once.
  rgcn_xw_wmma<<<782, 256, 2 * WPLANE_BYTES, stream>>>(feat, whi, wlo, xw);

  const int n4 = N_NODES_C * OUT_FEAT_C / 4;   // 1.6M float4
  rgcn_zero<<<(n4 + 255) / 256, 256, 0, stream>>>((float4*)y, n4);

  const unsigned scatter_blocks = (unsigned)(((size_t)N_EDGES_C * 64) / 256);
  rgcn_scatter<<<scatter_blocks, 256, 0, stream>>>(xw, esrc, edst, etyp, y);
}
